// GaudiMixtralSparseMoeBlock_37735582662897
// MI455X (gfx1250) — compile-verified, matrix operand order checked by reading, not run on hardware
//
#include <hip/hip_runtime.h>
#include <hip/hip_bf16.h>

// ---------------------------------------------------------------------------
// MoE (Mixtral-style) for gfx1250 / MI455X.
//  T=4096 tokens, H=1024, F=3584, E=8, top-2.
//  d_out = final [T*H] f32  ++  router_logits [T*E] f32
//
//  Pipeline:
//   1) zero counters
//   2) router (wave/token, wave32 shfl reduce, softmax/top-2)
//   3) offsets prefix-sum, 4) scatter -> compact per-expert row lists
//   5) convert x,w1,w3,w2 to bf16 (one streaming pass)
//   6) GEMM1: h = silu(x*w1^T)*(x*w3^T)  [bf16 WMMA, async-LDS double buffer]
//   7) GEMM2: y = (h*w2^T)*routing_w -> per-(token,slot) buffer
//   8) combine 2 slots per token
// ---------------------------------------------------------------------------

#define T_TOK 4096
#define HDIM  1024
#define FDIM  3584
#define NEXP  8
#define MT    128    // M tile (8 waves x 16 rows), 256-thread blocks
#define NT    64     // N tile (4 x 16-col wmma blocks)
#define KT    32     // K tile (bf16 wmma K)

typedef __attribute__((ext_vector_type(16))) __bf16 v16bf;
typedef __attribute__((ext_vector_type(8)))  float  v8f;

union FragB { v16bf v; uint4 q[2]; };

__device__ __forceinline__ unsigned short f2bf(float f) {
    union { float f; unsigned u; } x; x.f = f;
    unsigned r = x.u + 0x7FFFu + ((x.u >> 16) & 1u);   // round-to-nearest-even
    return (unsigned short)(r >> 16);
}

// Generic LDS pointer -> DS byte address (low 32 bits of the LDS aperture).
__device__ __forceinline__ unsigned lds_off(const void* p) {
    return (unsigned)(size_t)p;
}

// CDNA5 async global->LDS DMA (ASYNCcnt-tracked), 16B per lane.
__device__ __forceinline__ void async_b128(unsigned lds_addr, const void* gaddr) {
    asm volatile("global_load_async_to_lds_b128 %0, %1, off"
                 :: "v"(lds_addr), "v"(gaddr) : "memory");
}
__device__ __forceinline__ void wait_async0() {
    asm volatile("s_wait_asynccnt 0x0" ::: "memory");
}

// ---------------------------------------------------------------------------
__global__ void k_zero(int* counts, int* cursors) {
    if (threadIdx.x < NEXP) { counts[threadIdx.x] = 0; cursors[threadIdx.x] = 0; }
}

// fp32 -> bf16 streaming conversion (float4 in, ushort4 out)
__global__ __launch_bounds__(256) void k_convert(
        const float* __restrict__ s, unsigned short* __restrict__ d, int n4)
{
    int i = blockIdx.x * 256 + threadIdx.x;
    if (i >= n4) return;
    float4 v = ((const float4*)s)[i];
    ushort4 o;
    o.x = f2bf(v.x); o.y = f2bf(v.y); o.z = f2bf(v.z); o.w = f2bf(v.w);
    ((ushort4*)d)[i] = o;
}

// One wave (32 lanes) per token: logits, softmax, top-2, renormalize.
__global__ __launch_bounds__(256) void k_router(
        const float* __restrict__ x, const float* __restrict__ gw,
        float* __restrict__ logits, int* __restrict__ topk_id,
        float* __restrict__ topk_w, int* __restrict__ counts)
{
    const int t    = blockIdx.x * 8 + (threadIdx.x >> 5);
    const int lane = threadIdx.x & 31;
    const float* xr = x + (size_t)t * HDIM;

    float acc[NEXP];
#pragma unroll
    for (int e = 0; e < NEXP; ++e) acc[e] = 0.f;
    for (int h = lane; h < HDIM; h += 32) {
        float xv = xr[h];
#pragma unroll
        for (int e = 0; e < NEXP; ++e) acc[e] += xv * gw[e * HDIM + h];
    }
#pragma unroll
    for (int e = 0; e < NEXP; ++e) {
#pragma unroll
        for (int off = 16; off > 0; off >>= 1)
            acc[e] += __shfl_xor(acc[e], off, 32);
    }
    if (lane == 0) {
        float m = acc[0];
#pragma unroll
        for (int e = 1; e < NEXP; ++e) m = fmaxf(m, acc[e]);
        float p[NEXP]; float s = 0.f;
#pragma unroll
        for (int e = 0; e < NEXP; ++e) { p[e] = __expf(acc[e] - m); s += p[e]; }
        float inv = 1.f / s;
#pragma unroll
        for (int e = 0; e < NEXP; ++e) { p[e] *= inv; logits[t * NEXP + e] = acc[e]; }

        int i1 = 0; float p1 = p[0];
#pragma unroll
        for (int e = 1; e < NEXP; ++e) if (p[e] > p1) { p1 = p[e]; i1 = e; }
        int i2 = (i1 == 0) ? 1 : 0; float p2 = p[i2];
#pragma unroll
        for (int e = 0; e < NEXP; ++e) if (e != i1 && p[e] > p2) { p2 = p[e]; i2 = e; }

        float w = 1.f / (p1 + p2);
        topk_id[2 * t]     = i1;      topk_id[2 * t + 1] = i2;
        topk_w[2 * t]      = p1 * w;  topk_w[2 * t + 1]  = p2 * w;
        atomicAdd(&counts[i1], 1);    atomicAdd(&counts[i2], 1);
    }
}

__global__ void k_offsets(const int* __restrict__ counts, int* __restrict__ offs) {
    if (threadIdx.x == 0) {
        int o = 0;
        for (int e = 0; e < NEXP; ++e) { offs[e] = o; o += counts[e]; }
    }
}

__global__ __launch_bounds__(256) void k_scatter(
        const int* __restrict__ topk_id, const float* __restrict__ topk_w,
        const int* __restrict__ offs, int* __restrict__ cursors,
        int* __restrict__ rowtok, int* __restrict__ rowdst, float* __restrict__ roww)
{
    const int t = blockIdx.x * 256 + threadIdx.x;
    if (t >= T_TOK) return;
#pragma unroll
    for (int s = 0; s < 2; ++s) {
        int e   = topk_id[2 * t + s];
        int pos = offs[e] + atomicAdd(&cursors[e], 1);
        rowtok[pos] = t;
        rowdst[pos] = 2 * t + s;
        roww[pos]   = topk_w[2 * t + s];
    }
}

// ---------------------------------------------------------------------------
// GEMM1: gathered rows; h = silu(x*w1^T) * (x*w3^T) -> bf16 h_buf.
// 256 threads = 8 waves; wave w owns rows [16w, 16w+16) of the 128-row tile.
// All tiles filled by GLOBAL_LOAD_ASYNC_TO_LDS_B128, double buffered.
__global__ __launch_bounds__(256) void k_gemm1(
        const unsigned short* __restrict__ xb,  const unsigned short* __restrict__ w1b,
        const unsigned short* __restrict__ w3b, const int* __restrict__ counts,
        const int* __restrict__ offs, const int* __restrict__ rowtok,
        unsigned short* __restrict__ h_buf)
{
    const int e   = blockIdx.z;
    const int cnt = counts[e];
    const int mt  = blockIdx.x;
    if (mt * MT >= cnt) return;
    const int seg = offs[e];
    const int n0  = blockIdx.y * NT;

    __shared__ unsigned short lA [2][MT * KT];   // 2 x 8 KB
    __shared__ unsigned short lB1[2][NT * KT];   // 2 x 4 KB
    __shared__ unsigned short lB3[2][NT * KT];   // 2 x 4 KB
    __shared__ int stok[MT];

    const int tid  = threadIdx.x;
    const int wave = tid >> 5, lane = tid & 31;

    if (tid < MT) {
        int lr = mt * MT + tid;
        stok[tid] = rowtok[seg + (lr < cnt ? lr : cnt - 1)];
    }
    __syncthreads();

    v8f acc1[4], acc3[4];
    const v8f vz = {0.f,0.f,0.f,0.f,0.f,0.f,0.f,0.f};
#pragma unroll
    for (int nb = 0; nb < 4; ++nb) { acc1[nb] = vz; acc3[nb] = vz; }

    // A loader: 128 rows x 32 k; thread -> (row = tid>>1, 16-elem half)
    const int arow = tid >> 1;
    const int akh  = (tid & 1) * 16;
    const unsigned short* ax = xb + (size_t)stok[arow] * HDIM + akh;
    // B loaders: 64 rows x 32 k; thread -> (row = tid>>2, 8-elem quarter)
    const int brow = tid >> 2;
    const int bq   = (tid & 3) * 8;
    const unsigned short* b1p = w1b + ((size_t)e * FDIM + n0 + brow) * HDIM + bq;
    const unsigned short* b3p = w3b + ((size_t)e * FDIM + n0 + brow) * HDIM + bq;

    unsigned dA[2], dB1[2], dB3[2];
#pragma unroll
    for (int b = 0; b < 2; ++b) {
        dA[b]  = lds_off(&lA [b][arow * KT + akh]);
        dB1[b] = lds_off(&lB1[b][brow * KT + bq]);
        dB3[b] = lds_off(&lB3[b][brow * KT + bq]);
    }

    auto issue = [&](int b, int k) {    // 4 x b128 per thread
        async_b128(dA[b],       ax  + k);
        async_b128(dA[b]  + 16, ax  + k + 8);
        async_b128(dB1[b],      b1p + k);
        async_b128(dB3[b],      b3p + k);
    };
    issue(0, 0);                         // prologue: tile 0 -> buffer 0

    const int fr = lane & 15;
    const int fh = (lane >> 4) * 8;      // ISA 7.12.2 bf16 A/B lane layout

    int buf = 0;
    for (int k0 = 0; k0 < HDIM; k0 += KT, buf ^= 1) {
        wait_async0();                   // my tile-k0 DMA done
        __syncthreads();                 // everyone's done; prev buffer free
        if (k0 + KT < HDIM) issue(buf ^ 1, k0 + KT);   // overlap with compute

        FragB a;
        const unsigned short* ap = &lA[buf][(wave * 16 + fr) * KT];
        a.q[0] = *(const uint4*)(ap + fh);
        a.q[1] = *(const uint4*)(ap + 16 + fh);
        FragB bf1[4], bf3[4];
#pragma unroll
        for (int nb = 0; nb < 4; ++nb) {     // load all fragments first (ILP)
            const unsigned short* bp = &lB1[buf][(nb * 16 + fr) * KT];
            bf1[nb].q[0] = *(const uint4*)(bp + fh);
            bf1[nb].q[1] = *(const uint4*)(bp + 16 + fh);
            const unsigned short* cp = &lB3[buf][(nb * 16 + fr) * KT];
            bf3[nb].q[0] = *(const uint4*)(cp + fh);
            bf3[nb].q[1] = *(const uint4*)(cp + 16 + fh);
        }
#pragma unroll
        for (int nb = 0; nb < 4; ++nb) {     // 8 back-to-back wmma
            acc1[nb] = __builtin_amdgcn_wmma_f32_16x16x32_bf16(
                false, a.v, false, bf1[nb].v, (short)0, acc1[nb], false, false);
            acc3[nb] = __builtin_amdgcn_wmma_f32_16x16x32_bf16(
                false, a.v, false, bf3[nb].v, (short)0, acc3[nb], false, false);
        }
    }

    // Epilogue: SwiGLU, bf16 store. C layout: VGPR r <-> M = r + 8*(lane>=16)
    const int lrow = wave * 16 + ((lane >> 4) << 3);
    const int gr0  = mt * MT + lrow;
    const int col0 = n0 + (lane & 15);
#pragma unroll
    for (int r = 0; r < 8; ++r) {
        int gr = gr0 + r;
        if (gr < cnt) {
#pragma unroll
            for (int nb = 0; nb < 4; ++nb) {
                float a1v = acc1[nb][r], a3v = acc3[nb][r];
                float hv = (a1v / (1.f + __expf(-a1v))) * a3v;
                h_buf[(size_t)(seg + gr) * FDIM + col0 + nb * 16] = f2bf(hv);
            }
        }
    }
}

// ---------------------------------------------------------------------------
// GEMM2: y = h * w2^T (K = F), scaled by routing weight, scattered to y_buf
// row dst = 2*token + slot (deterministic, atomic-free).
__global__ __launch_bounds__(256) void k_gemm2(
        const unsigned short* __restrict__ h_buf, const unsigned short* __restrict__ w2b,
        const int* __restrict__ counts, const int* __restrict__ offs,
        const int* __restrict__ rowdst, const float* __restrict__ roww,
        float* __restrict__ y_buf)
{
    const int e   = blockIdx.z;
    const int cnt = counts[e];
    const int mt  = blockIdx.x;
    if (mt * MT >= cnt) return;
    const int seg = offs[e];
    const int n0  = blockIdx.y * NT;

    __shared__ unsigned short lA[2][MT * KT];    // 2 x 8 KB
    __shared__ unsigned short lB[2][NT * KT];    // 2 x 4 KB
    __shared__ int   sdst[MT];
    __shared__ float sw  [MT];

    const int tid = threadIdx.x, wave = tid >> 5, lane = tid & 31;
    if (tid < MT) {
        int lr = mt * MT + tid;
        int rr = lr < cnt ? lr : cnt - 1;
        sdst[tid] = rowdst[seg + rr];
        sw[tid]   = roww[seg + rr];
    }

    v8f acc[4];
    const v8f vz = {0.f,0.f,0.f,0.f,0.f,0.f,0.f,0.f};
#pragma unroll
    for (int nb = 0; nb < 4; ++nb) acc[nb] = vz;

    const int arow = tid >> 1;
    const int akh  = (tid & 1) * 16;
    const int lrA  = mt * MT + arow;
    const int rrA  = lrA < cnt ? lrA : cnt - 1;          // clamp garbage rows
    const unsigned short* hrow = h_buf + (size_t)(seg + rrA) * FDIM + akh;
    const int brow = tid >> 2;
    const int bq   = (tid & 3) * 8;
    const unsigned short* w2p = w2b + ((size_t)e * HDIM + n0 + brow) * FDIM + bq;

    unsigned dA[2], dB[2];
#pragma unroll
    for (int b = 0; b < 2; ++b) {
        dA[b] = lds_off(&lA[b][arow * KT + akh]);
        dB[b] = lds_off(&lB[b][brow * KT + bq]);
    }
    auto issue = [&](int b, int k) {    // 3 x b128 per thread
        async_b128(dA[b],      hrow + k);
        async_b128(dA[b] + 16, hrow + k + 8);
        async_b128(dB[b],      w2p + k);
    };
    issue(0, 0);

    const int fr = lane & 15;
    const int fh = (lane >> 4) * 8;

    int buf = 0;
    for (int k0 = 0; k0 < FDIM; k0 += KT, buf ^= 1) {
        wait_async0();
        __syncthreads();
        if (k0 + KT < FDIM) issue(buf ^ 1, k0 + KT);

        FragB a;
        const unsigned short* ap = &lA[buf][(wave * 16 + fr) * KT];
        a.q[0] = *(const uint4*)(ap + fh);
        a.q[1] = *(const uint4*)(ap + 16 + fh);
        FragB bfr[4];
#pragma unroll
        for (int nb = 0; nb < 4; ++nb) {
            const unsigned short* bp = &lB[buf][(nb * 16 + fr) * KT];
            bfr[nb].q[0] = *(const uint4*)(bp + fh);
            bfr[nb].q[1] = *(const uint4*)(bp + 16 + fh);
        }
#pragma unroll
        for (int nb = 0; nb < 4; ++nb) {
            acc[nb] = __builtin_amdgcn_wmma_f32_16x16x32_bf16(
                false, a.v, false, bfr[nb].v, (short)0, acc[nb], false, false);
        }
    }

    const int lrow = wave * 16 + ((lane >> 4) << 3);
    const int gr0  = mt * MT + lrow;
    const int colb = n0 + (lane & 15);
#pragma unroll
    for (int r = 0; r < 8; ++r) {
        int gr = gr0 + r;
        if (gr < cnt) {
            int   dst = sdst[lrow + r];
            float wgt = sw  [lrow + r];
#pragma unroll
            for (int nb = 0; nb < 4; ++nb)
                y_buf[(size_t)dst * HDIM + colb + nb * 16] = acc[nb][r] * wgt;
        }
    }
}

// final[t][c] = y_buf[2t][c] + y_buf[2t+1][c]
__global__ __launch_bounds__(256) void k_combine(
        const float* __restrict__ y_buf, float* __restrict__ out)
{
    size_t g = (size_t)blockIdx.x * 256 + threadIdx.x;
    size_t t = g >> 10;            // H == 1024
    size_t c = g & 1023;
    out[g] = y_buf[(t * 2) * HDIM + c] + y_buf[(t * 2 + 1) * HDIM + c];
}

// ---------------------------------------------------------------------------
extern "C" void kernel_launch(void* const* d_in, const int* in_sizes, int n_in,
                              void* d_out, int out_size, void* d_ws, size_t ws_size,
                              hipStream_t stream)
{
    (void)in_sizes; (void)n_in; (void)out_size; (void)ws_size;
    const float* x  = (const float*)d_in[0];   // [T,H]
    const float* gw = (const float*)d_in[1];   // [E,H]
    const float* w1 = (const float*)d_in[2];   // [E,F,H]
    const float* w2 = (const float*)d_in[3];   // [E,H,F]
    const float* w3 = (const float*)d_in[4];   // [E,F,H]
    float* out    = (float*)d_out;
    float* logits = out + (size_t)T_TOK * HDIM;

    // Workspace carve-up (~277 MB total)
    char* p = (char*)d_ws;
    int*   counts  = (int*)p;            p += 256;
    int*   cursors = (int*)p;            p += 256;
    int*   offs    = (int*)p;            p += 256;
    int*   topk_id = (int*)p;            p += (size_t)2 * T_TOK * sizeof(int);
    float* topk_w  = (float*)p;          p += (size_t)2 * T_TOK * sizeof(float);
    int*   rowtok  = (int*)p;            p += (size_t)2 * T_TOK * sizeof(int);
    int*   rowdst  = (int*)p;            p += (size_t)2 * T_TOK * sizeof(int);
    float* roww    = (float*)p;          p += (size_t)2 * T_TOK * sizeof(float);
    unsigned short* xb  = (unsigned short*)p; p += (size_t)T_TOK * HDIM * 2;         // 8.4 MB
    unsigned short* w1b = (unsigned short*)p; p += (size_t)NEXP * FDIM * HDIM * 2;   // 58.7 MB
    unsigned short* w3b = (unsigned short*)p; p += (size_t)NEXP * FDIM * HDIM * 2;   // 58.7 MB
    unsigned short* w2b = (unsigned short*)p; p += (size_t)NEXP * HDIM * FDIM * 2;   // 58.7 MB
    unsigned short* h_buf = (unsigned short*)p; p += (size_t)2 * T_TOK * FDIM * 2;   // 58.7 MB
    float* y_buf = (float*)p;                                                        // 33.6 MB

    const int nx4 = (T_TOK * HDIM) / 4;
    const int nw4 = (NEXP * FDIM * HDIM) / 4;

    hipLaunchKernelGGL(k_zero,    dim3(1),           dim3(32),  0, stream, counts, cursors);
    hipLaunchKernelGGL(k_router,  dim3(T_TOK / 8),   dim3(256), 0, stream,
                       x, gw, logits, topk_id, topk_w, counts);
    hipLaunchKernelGGL(k_offsets, dim3(1),           dim3(32),  0, stream, counts, offs);
    hipLaunchKernelGGL(k_scatter, dim3(T_TOK / 256), dim3(256), 0, stream,
                       topk_id, topk_w, offs, cursors, rowtok, rowdst, roww);

    hipLaunchKernelGGL(k_convert, dim3((nx4 + 255) / 256), dim3(256), 0, stream, x,  xb,  nx4);
    hipLaunchKernelGGL(k_convert, dim3((nw4 + 255) / 256), dim3(256), 0, stream, w1, w1b, nw4);
    hipLaunchKernelGGL(k_convert, dim3((nw4 + 255) / 256), dim3(256), 0, stream, w3, w3b, nw4);
    hipLaunchKernelGGL(k_convert, dim3((nw4 + 255) / 256), dim3(256), 0, stream, w2, w2b, nw4);

    hipLaunchKernelGGL(k_gemm1, dim3(T_TOK / MT, FDIM / NT, NEXP), dim3(256), 0, stream,
                       xb, w1b, w3b, counts, offs, rowtok, h_buf);
    hipLaunchKernelGGL(k_gemm2, dim3(T_TOK / MT, HDIM / NT, NEXP), dim3(256), 0, stream,
                       h_buf, w2b, counts, offs, rowdst, roww, y_buf);
    hipLaunchKernelGGL(k_combine, dim3((T_TOK * HDIM) / 256), dim3(256), 0, stream,
                       y_buf, out);
}
